// MoeLayer_1906965480028
// MI455X (gfx1250) — matrix operand order repaired and checked by reading, not compiled, not run on hardware
//
#include <hip/hip_runtime.h>

// ---------------- problem constants ----------------
constexpr int EN = 8;      // experts
constexpr int HD = 1024;   // hidden
constexpr int FD = 2048;   // d_ff
constexpr int TT = 4096;   // tokens (B*S)
constexpr int MT = 64;     // M tile (tokens per block)

typedef __bf16 bf16;
typedef __attribute__((ext_vector_type(16))) __bf16 v16bf;
typedef __attribute__((ext_vector_type(8)))  float  v8f;
typedef __attribute__((ext_vector_type(4)))  int    v4i;

union Frag16 { v16bf v; uint4 q[2]; bf16 h[16]; };
union Frag8  { uint4 q; bf16 h[8]; };

// ---- CDNA5 async global->LDS copy (ASYNCcnt-tracked), guarded by __has_builtin ----
#if defined(__has_builtin)
#if __has_builtin(__builtin_amdgcn_global_load_async_to_lds_b128)
#define HAS_ASYNC_LDS 1
#endif
#endif

#define GLOBAL_AS __attribute__((address_space(1)))
#define LDS_AS    __attribute__((address_space(3)))

__device__ __forceinline__ void async_copy16(const bf16* g, bf16* l) {
#ifdef HAS_ASYNC_LDS
  __builtin_amdgcn_global_load_async_to_lds_b128(
      (GLOBAL_AS v4i*)g,
      (LDS_AS v4i*)l, 0, 0);
#else
  *(uint4*)l = *(const uint4*)g;
#endif
}

__device__ __forceinline__ void wait_async_and_barrier() {
#ifdef HAS_ASYNC_LDS
#if __has_builtin(__builtin_amdgcn_s_wait_asynccnt)
  __builtin_amdgcn_s_wait_asynccnt(0);
#else
  asm volatile("s_wait_asynccnt 0x0" ::: "memory");
#endif
#endif
  __syncthreads();
}

// A-fragment (16x32 bf16, M x K) from LDS tile stored row-major, ld = 40 halves.
// Contiguous k order used here (matches B pack): lanes 0-15 -> M=lane, K = {0..15}; lanes 16-31 -> M=lane-16, K = {16..31}
__device__ __forceinline__ v16bf lds_a_frag(const bf16* sA, int lane) {
  const bf16* p = sA + (lane & 15) * 40 + ((lane >> 4) & 1) * 16;
  Frag16 f;
  f.q[0] = *(const uint4*)(p);
  f.q[1] = *(const uint4*)(p + 8);
  return f.v;
}

// B-fragment from pre-swizzled global memory: 1KB block, lane reads 32B at lane*16 halves.
__device__ __forceinline__ v16bf ldg_b_frag(const bf16* fragBase, int lane) {
  return *(const v16bf*)(fragBase + lane * 16);
}

__device__ __forceinline__ v8f vzero8() { v8f z = {0.f,0.f,0.f,0.f,0.f,0.f,0.f,0.f}; return z; }

#define WMMA_BF16(A, B, C) \
  __builtin_amdgcn_wmma_f32_16x16x32_bf16(false, (A), false, (B), (short)0, (C), false, false)

// ---------------- kernel 0: init ----------------
__global__ void moe_init_kernel(float* __restrict__ out, int n, int* __restrict__ counts) {
  int i = blockIdx.x * blockDim.x + threadIdx.x;
  if (i < n) out[i] = 0.0f;
  if (i < EN) counts[i] = 0;
}

// ---------------- kernel 0b: x -> bf16 ----------------
__global__ void moe_xconv_kernel(const float* __restrict__ x, bf16* __restrict__ xb) {
  int i = blockIdx.x * blockDim.x + threadIdx.x;   // 8 floats per thread
  size_t o = (size_t)i * 8;
  if (o < (size_t)TT * HD) {
    float4 f0 = *(const float4*)(x + o);
    float4 f1 = *(const float4*)(x + o + 4);
    Frag8 h8;
    h8.h[0] = (bf16)f0.x; h8.h[1] = (bf16)f0.y; h8.h[2] = (bf16)f0.z; h8.h[3] = (bf16)f0.w;
    h8.h[4] = (bf16)f1.x; h8.h[5] = (bf16)f1.y; h8.h[6] = (bf16)f1.z; h8.h[7] = (bf16)f1.w;
    *(uint4*)(xb + o) = h8.q;
  }
}

// ---------------- kernel 1: routing (top-2 softmax + expert lists) ----------------
__global__ void __launch_bounds__(256) moe_route_kernel(
    const float* __restrict__ x, const float* __restrict__ Wgate, const float* __restrict__ bgate,
    int* __restrict__ counts, int* __restrict__ listTok, float* __restrict__ listW) {
  const int t = blockIdx.x;
  const int tid = threadIdx.x;
  float acc[EN];
#pragma unroll
  for (int e = 0; e < EN; e++) acc[e] = 0.0f;
  const float* xr = x + (size_t)t * HD;
  for (int h = tid; h < HD; h += 256) {
    float xv = xr[h];
    const float* wr = Wgate + (size_t)h * EN;
#pragma unroll
    for (int e = 0; e < EN; e++) acc[e] += xv * wr[e];
  }
  __shared__ float red[256];
  __shared__ float logits[EN];
  for (int e = 0; e < EN; e++) {
    red[tid] = acc[e];
    __syncthreads();
    for (int s = 128; s > 0; s >>= 1) {
      if (tid < s) red[tid] += red[tid + s];
      __syncthreads();
    }
    if (tid == 0) logits[e] = red[0] + bgate[e];
    __syncthreads();
  }
  if (tid == 0) {
    int i0 = 0; float v0 = logits[0];
    for (int e = 1; e < EN; e++) if (logits[e] > v0) { v0 = logits[e]; i0 = e; }
    int i1 = (i0 == 0) ? 1 : 0; float v1 = logits[i1];
    for (int e = 0; e < EN; e++) if (e != i0 && logits[e] > v1) { v1 = logits[e]; i1 = e; }
    float e0 = __expf(v0 - v0), e1 = __expf(v1 - v0);
    float inv = 1.0f / (e0 + e1);
    int s0 = atomicAdd(&counts[i0], 1);
    listTok[i0 * TT + s0] = t; listW[i0 * TT + s0] = e0 * inv;
    int s1 = atomicAdd(&counts[i1], 1);
    listTok[i1 * TT + s1] = t; listW[i1 * TT + s1] = e1 * inv;
  }
}

// ---------------- kernel 2: exclusive scan of counts ----------------
__global__ void moe_scan_kernel(const int* __restrict__ counts, int* __restrict__ base) {
  if (threadIdx.x == 0) {
    int b = 0;
    for (int e = 0; e < EN; e++) { base[e] = b; b += counts[e]; }
  }
}

// ---------------- kernel 3: fp32 -> bf16 weight conversion into WMMA-B fragment layout ----
// src: [E][Kd][N] row-major fp32. dst: frag blocks of 512 bf16;
// fragIdx = (e*ntiles + nt)*ktiles + kt; within block: lane*16 + s,
// element (k = kt*32 + (lane&16) + s, col = nt*16 + lane%16).
__global__ void __launch_bounds__(256) moe_convert_kernel(
    const float* __restrict__ src, bf16* __restrict__ dst, int Kd, int N, int nFrag) {
  const int lane = threadIdx.x & 31;
  const int wave = (blockIdx.x * blockDim.x + threadIdx.x) >> 5;
  const int nWaves = (gridDim.x * blockDim.x) >> 5;
  const int ktiles = Kd >> 5, ntiles = N >> 4;
  for (int f = wave; f < nFrag; f += nWaves) {
    int kt = f % ktiles;
    int rest = f / ktiles;
    int nt = rest % ntiles;
    int e = rest / ntiles;
    const float* S = src + (size_t)e * Kd * N;
    int col = nt * 16 + (lane & 15);
    int k0 = kt * 32 + (lane & 16);
    Frag16 fr;
#pragma unroll
    for (int s = 0; s < 16; s++) fr.h[s] = (bf16)S[(size_t)(k0 + s) * N + col];
    *(v16bf*)(dst + (size_t)f * 512 + lane * 16) = fr.v;
  }
}

// ---------------- kernel 4: z = (x@W1 + b1) * (x@W2 + b2), gathered rows -> bf16 ------
__global__ void __launch_bounds__(256) moe_z_kernel(
    const bf16* __restrict__ xb, const bf16* __restrict__ W1b, const bf16* __restrict__ W2b,
    const float* __restrict__ b1, const float* __restrict__ b2,
    const int* __restrict__ listTok, const int* __restrict__ counts, const int* __restrict__ base,
    bf16* __restrict__ z) {
  const int e = blockIdx.z;
  const int count = counts[e];
  const int tileBase = blockIdx.x * MT;
  if (tileBase >= count) return;
  const int tid = threadIdx.x, lane = tid & 31, wave = tid >> 5;
  const int mSub = wave & 3, nGrp = wave >> 2;
  const int n0 = blockIdx.y * 128 + nGrp * 64;

  __shared__ __align__(16) bf16 sA[2][MT * 40];

  const int sr = tid >> 2, sc = (tid & 3) * 8;
  int gi = tileBase + sr; if (gi > count - 1) gi = count - 1;
  const bf16* arow = xb + (size_t)listTok[e * TT + gi] * HD;

  v8f a1[4], a2[4];
#pragma unroll
  for (int i = 0; i < 4; i++) { a1[i] = vzero8(); a2[i] = vzero8(); }

  const int ktiles = HD / 32;                       // 32
  const size_t matStride = (size_t)(FD / 16) * ktiles * 512;
  const bf16* W1e = W1b + (size_t)e * matStride;
  const bf16* W2e = W2b + (size_t)e * matStride;

  async_copy16(arow + sc, &sA[0][sr * 40 + sc]);
  wait_async_and_barrier();

  for (int kt = 0; kt < ktiles; kt++) {
    const int cur = kt & 1;
    if (kt + 1 < ktiles)
      async_copy16(arow + (kt + 1) * 32 + sc, &sA[cur ^ 1][sr * 40 + sc]);

    v16bf bf[8];
    const size_t ko = (size_t)kt * 512;
#pragma unroll
    for (int nf = 0; nf < 4; nf++) {
      size_t off = (size_t)((n0 >> 4) + nf) * ktiles * 512 + ko;
      bf[nf * 2]     = ldg_b_frag(W1e + off, lane);
      bf[nf * 2 + 1] = ldg_b_frag(W2e + off, lane);
    }
    v16bf afrag = lds_a_frag(&sA[cur][mSub * 16 * 40], lane);
#pragma unroll
    for (int nf = 0; nf < 4; nf++) {
      a1[nf] = WMMA_BF16(afrag, bf[nf * 2], a1[nf]);
      a2[nf] = WMMA_BF16(afrag, bf[nf * 2 + 1], a2[nf]);
    }
    wait_async_and_barrier();
  }
  const int rl = mSub * 16 + ((lane >> 4) & 1) * 8;
  const size_t rowG0 = (size_t)(base[e] + tileBase);
#pragma unroll
  for (int nf = 0; nf < 4; nf++) {
    int col = n0 + nf * 16 + (lane & 15);
    float bb1 = b1[e * FD + col], bb2 = b2[e * FD + col];
#pragma unroll
    for (int r = 0; r < 8; r++) {
      float v = (a1[nf][r] + bb1) * (a2[nf][r] + bb2);
      z[(rowG0 + rl + r) * (size_t)FD + col] = (bf16)v;
    }
  }
}

// ---------------- kernel 5: o = silu(z@Wg + bg) * (z@Wl + bl) -> bf16 ----------------
__global__ void __launch_bounds__(256) moe_gl_kernel(
    const bf16* __restrict__ z, const bf16* __restrict__ Wgb, const bf16* __restrict__ Wlb,
    const float* __restrict__ bg, const float* __restrict__ bl,
    const int* __restrict__ counts, const int* __restrict__ base,
    bf16* __restrict__ o) {
  const int e = blockIdx.z;
  const int count = counts[e];
  const int tileBase = blockIdx.x * MT;
  if (tileBase >= count) return;
  const int tid = threadIdx.x, lane = tid & 31, wave = tid >> 5;
  const int mSub = wave & 3, nGrp = wave >> 2;
  const int n0 = blockIdx.y * 128 + nGrp * 64;

  __shared__ __align__(16) bf16 sA[2][MT * 40];
  const int sr = tid >> 2, sc = (tid & 3) * 8;
  const size_t rowG0 = (size_t)(base[e] + tileBase);
  const bf16* arow = z + (rowG0 + sr) * (size_t)FD;

  v8f ag[4], al[4];
#pragma unroll
  for (int i = 0; i < 4; i++) { ag[i] = vzero8(); al[i] = vzero8(); }

  const int ktiles = FD / 32;                        // 64
  const size_t matStride = (size_t)(FD / 16) * ktiles * 512;
  const bf16* Wge = Wgb + (size_t)e * matStride;
  const bf16* Wle = Wlb + (size_t)e * matStride;

  async_copy16(arow + sc, &sA[0][sr * 40 + sc]);
  wait_async_and_barrier();

  for (int kt = 0; kt < ktiles; kt++) {
    const int cur = kt & 1;
    if (kt + 1 < ktiles)
      async_copy16(arow + (kt + 1) * 32 + sc, &sA[cur ^ 1][sr * 40 + sc]);

    v16bf bf[8];
    const size_t ko = (size_t)kt * 512;
#pragma unroll
    for (int nf = 0; nf < 4; nf++) {
      size_t off = (size_t)((n0 >> 4) + nf) * ktiles * 512 + ko;
      bf[nf * 2]     = ldg_b_frag(Wge + off, lane);
      bf[nf * 2 + 1] = ldg_b_frag(Wle + off, lane);
    }
    v16bf afrag = lds_a_frag(&sA[cur][mSub * 16 * 40], lane);
#pragma unroll
    for (int nf = 0; nf < 4; nf++) {
      ag[nf] = WMMA_BF16(afrag, bf[nf * 2], ag[nf]);
      al[nf] = WMMA_BF16(afrag, bf[nf * 2 + 1], al[nf]);
    }
    wait_async_and_barrier();
  }
  const int rl = mSub * 16 + ((lane >> 4) & 1) * 8;
#pragma unroll
  for (int nf = 0; nf < 4; nf++) {
    int col = n0 + nf * 16 + (lane & 15);
    float bbg = bg[e * FD + col], bbl = bl[e * FD + col];
#pragma unroll
    for (int r = 0; r < 8; r++) {
      float g = ag[nf][r] + bbg;
      float l = al[nf][r] + bbl;
      float sw = g * (1.0f / (1.0f + __expf(-g)));   // silu(g)
      o[(rowG0 + rl + r) * (size_t)FD + col] = (bf16)(sw * l);
    }
  }
}

// ---------------- kernel 6: y = o@W3 + b3, scaled by gate weight, atomicAdd to out ----
__global__ void __launch_bounds__(256) moe_y_kernel(
    const bf16* __restrict__ o, const bf16* __restrict__ W3b, const float* __restrict__ b3,
    const int* __restrict__ listTok, const float* __restrict__ listW,
    const int* __restrict__ counts, const int* __restrict__ base,
    float* __restrict__ out) {
  const int e = blockIdx.z;
  const int count = counts[e];
  const int tileBase = blockIdx.x * MT;
  if (tileBase >= count) return;
  const int tid = threadIdx.x, lane = tid & 31, wave = tid >> 5;
  const int mSub = wave & 3, nGrp = wave >> 2;
  const int n0 = blockIdx.y * 128 + nGrp * 64;

  __shared__ __align__(16) bf16 sA[2][MT * 40];
  const int sr = tid >> 2, sc = (tid & 3) * 8;
  const size_t rowG0 = (size_t)(base[e] + tileBase);
  const bf16* arow = o + (rowG0 + sr) * (size_t)FD;

  v8f ay[4];
#pragma unroll
  for (int i = 0; i < 4; i++) ay[i] = vzero8();

  const int ktiles = FD / 32;                        // 64
  const size_t matStride = (size_t)(HD / 16) * ktiles * 512;
  const bf16* W3e = W3b + (size_t)e * matStride;

  async_copy16(arow + sc, &sA[0][sr * 40 + sc]);
  wait_async_and_barrier();

  for (int kt = 0; kt < ktiles; kt++) {
    const int cur = kt & 1;
    if (kt + 1 < ktiles)
      async_copy16(arow + (kt + 1) * 32 + sc, &sA[cur ^ 1][sr * 40 + sc]);

    v16bf bf[4];
    const size_t ko = (size_t)kt * 512;
#pragma unroll
    for (int nf = 0; nf < 4; nf++) {
      size_t off = (size_t)((n0 >> 4) + nf) * ktiles * 512 + ko;
      bf[nf] = ldg_b_frag(W3e + off, lane);
    }
    v16bf afrag = lds_a_frag(&sA[cur][mSub * 16 * 40], lane);
#pragma unroll
    for (int nf = 0; nf < 4; nf++) {
      ay[nf] = WMMA_BF16(afrag, bf[nf], ay[nf]);
    }
    wait_async_and_barrier();
  }
  const int rl = mSub * 16 + ((lane >> 4) & 1) * 8;
  int tokR[8]; float wR[8]; bool okR[8];
#pragma unroll
  for (int r = 0; r < 8; r++) {
    int i = tileBase + rl + r;
    okR[r] = (i < count);
    int ic = okR[r] ? i : count - 1;
    tokR[r] = listTok[e * TT + ic];
    wR[r] = listW[e * TT + ic];
  }
#pragma unroll
  for (int nf = 0; nf < 4; nf++) {
    int col = n0 + nf * 16 + (lane & 15);
    float bb = b3[e * HD + col];
#pragma unroll
    for (int r = 0; r < 8; r++) {
      float v = (ay[nf][r] + bb) * wR[r];
      if (okR[r]) atomicAdd(out + (size_t)tokR[r] * HD + col, v);
    }
  }
}

// ---------------- host launcher ----------------
extern "C" void kernel_launch(void* const* d_in, const int* in_sizes, int n_in,
                              void* d_out, int out_size, void* d_ws, size_t ws_size,
                              hipStream_t stream) {
  const float* x     = (const float*)d_in[0];
  const float* Wgate = (const float*)d_in[1];
  const float* bgate = (const float*)d_in[2];
  const float* W1    = (const float*)d_in[3];
  const float* b1    = (const float*)d_in[4];
  const float* W2    = (const float*)d_in[5];
  const float* b2    = (const float*)d_in[6];
  const float* Wg    = (const float*)d_in[7];
  const float* bg    = (const float*)d_in[8];
  const float* Wl    = (const float*)d_in[9];
  const float* bl    = (const float*)d_in[10];
  const float* W3    = (const float*)d_in[11];
  const float* b3    = (const float*)d_in[12];
  float* out = (float*)d_out;

  char* p = (char*)d_ws;
  auto carve = [&](size_t bytes) -> char* {
    char* r = p;
    p += (bytes + 255) & ~(size_t)255;
    return r;
  };
  bf16* W1b = (bf16*)carve((size_t)EN * HD * FD * 2);
  bf16* W2b = (bf16*)carve((size_t)EN * HD * FD * 2);
  bf16* Wgb = (bf16*)carve((size_t)EN * FD * FD * 2);
  bf16* Wlb = (bf16*)carve((size_t)EN * FD * FD * 2);
  bf16* W3b = (bf16*)carve((size_t)EN * FD * HD * 2);
  bf16* zB  = (bf16*)carve((size_t)2 * TT * FD * 2);
  bf16* oB  = (bf16*)carve((size_t)2 * TT * FD * 2);
  bf16* xb  = (bf16*)carve((size_t)TT * HD * 2);
  int*   listTok = (int*)carve((size_t)EN * TT * 4);
  float* listW   = (float*)carve((size_t)EN * TT * 4);
  int*   counts  = (int*)carve(EN * 4);
  int*   baseOff = (int*)carve(EN * 4);

  // 0) zero output + counters; convert x to bf16
  {
    int n = out_size;
    moe_init_kernel<<<(n + 255) / 256, 256, 0, stream>>>(out, n, counts);
  }
  moe_xconv_kernel<<<(TT * HD / 8 + 255) / 256, 256, 0, stream>>>(x, xb);
  // 1) routing
  moe_route_kernel<<<TT, 256, 0, stream>>>(x, Wgate, bgate, counts, listTok, listW);
  // 2) scan
  moe_scan_kernel<<<1, 32, 0, stream>>>(counts, baseOff);
  // 3) weight conversion (bf16, WMMA-B fragment layout)
  moe_convert_kernel<<<4096, 256, 0, stream>>>(W1, W1b, HD, FD, EN * (HD / 32) * (FD / 16));
  moe_convert_kernel<<<4096, 256, 0, stream>>>(W2, W2b, HD, FD, EN * (HD / 32) * (FD / 16));
  moe_convert_kernel<<<4096, 256, 0, stream>>>(Wg, Wgb, FD, FD, EN * (FD / 32) * (FD / 16));
  moe_convert_kernel<<<4096, 256, 0, stream>>>(Wl, Wlb, FD, FD, EN * (FD / 32) * (FD / 16));
  moe_convert_kernel<<<4096, 256, 0, stream>>>(W3, W3b, FD, HD, EN * (FD / 32) * (HD / 16));
  // 4) z = (xW1+b1)*(xW2+b2)
  moe_z_kernel<<<dim3(TT / MT, FD / 128, EN), 256, 0, stream>>>(
      xb, W1b, W2b, b1, b2, listTok, counts, baseOff, zB);
  // 5) o = silu(zWg+bg)*(zWl+bl)
  moe_gl_kernel<<<dim3(TT / MT, FD / 128, EN), 256, 0, stream>>>(
      zB, Wgb, Wlb, bg, bl, counts, baseOff, oB);
  // 6) out += w * (oW3 + b3)
  moe_y_kernel<<<dim3(TT / MT, HD / 128, EN), 256, 0, stream>>>(
      oB, W3b, b3, listTok, listW, counts, baseOff, out);
}
